// CausalSelfAttention_77833397338836
// MI455X (gfx1250) — matrix operand order repaired and checked by reading, not compiled, not run on hardware
//
#include <hip/hip_runtime.h>

typedef __bf16 bf16_t;
typedef __attribute__((ext_vector_type(8)))  bf16_t v8bf;
typedef __attribute__((ext_vector_type(16))) bf16_t v16bf;
typedef __attribute__((ext_vector_type(8)))  float  v8f;
typedef __attribute__((ext_vector_type(4)))  unsigned u32x4;
typedef __attribute__((ext_vector_type(8)))  unsigned u32x8;

#define D_MODEL   1024
#define NUM_HEADS 16
#define HEAD_DIM  64
#define BATCH     4
#define SEQ       2048
#define MTOT      (BATCH * SEQ)
#define LDS_STRIDE 72   // 64 + 8 halves pad (144B rows); matches TDM pad settings

// ---------- helpers ----------

__device__ __forceinline__ unsigned short f2bf_bits(float f) {
  unsigned int u = __float_as_uint(f);
  unsigned int r = u + 0x7fffu + ((u >> 16) & 1u);   // round-to-nearest-even
  return (unsigned short)(r >> 16);
}

__device__ __forceinline__ unsigned lds_off(const void* p) {
  // generic pointers into LDS carry the ds offset in the low 32 bits
  return (unsigned)(uintptr_t)p;
}

// Tensor Data Mover: DMA one 64x64 bf16 tile (2D tensor) global -> LDS.
// LDS rows get 16B padding every 128B (pad_interval=4 -> 32 dwords,
// pad_amount=3 -> 4 dwords) => row stride 144B = LDS_STRIDE halves.
// D# group0/group1 packed per CDNA5 ISA 8.3/8.4; 2-group form (<=2D tensor).
__device__ __forceinline__ void tdm_load_tile64x64(
    unsigned lds_addr, const void* gaddr,
    unsigned row_elems /* tensor_dim0 == dim0 stride */, unsigned rows_total) {
  unsigned long long ga = (unsigned long long)(uintptr_t)gaddr;
  u32x4 g0;
  g0[0] = 1u;                                               // count=1, user D#
  g0[1] = lds_addr;                                         // lds_addr[31:0]
  g0[2] = (unsigned)ga;                                     // global_addr[31:0]
  g0[3] = (unsigned)((ga >> 32) & 0x01FFFFFFu) | (2u << 30);// ga[56:32] | type=2
  u32x8 g1;
  g1[0] = (1u << 16)        // data_size = 2 bytes
        | (1u << 20)        // pad_enable
        | (4u << 22)        // pad_interval: 32 dwords (128B)
        | (3u << 25);       // pad_amount: 4 dwords (16B)
  g1[1] = (row_elems & 0xFFFFu) << 16;                      // tensor_dim0[15:0]
  g1[2] = (row_elems >> 16) | ((rows_total & 0xFFFFu) << 16); // dim0 hi | dim1 lo
  g1[3] = (rows_total >> 16) | (64u << 16);                 // dim1 hi | tile_dim0
  g1[4] = 64u;                                              // tile_dim1 | tile_dim2=0
  g1[5] = row_elems;                                        // tensor_dim0_stride[31:0]
  g1[6] = 0u;                                               // stride0 hi | stride1 lo
  g1[7] = 0u;                                               // stride1 hi
  asm volatile("tensor_load_to_lds %0, %1" :: "s"(g0), "s"(g1) : "memory");
}

// A-matrix 16x32 bf16 fragment per ISA layout:
// lane L: row = L&15 ; K = (L>>4)*8 + (e&7) + (e>>3)*16
__device__ __forceinline__ v16bf load_a_frag(const bf16_t* base, int stride,
                                             int row0, int k0, int lane) {
  const bf16_t* p = base + (size_t)(row0 + (lane & 15)) * stride + k0 + ((lane >> 4) << 3);
  v8bf lo = *(const v8bf*)p;          // K = b .. b+7
  v8bf hi = *(const v8bf*)(p + 16);   // K = b+16 .. b+23
  v16bf r;
#pragma unroll
  for (int i = 0; i < 8; ++i) { r[i] = lo[i]; r[i + 8] = hi[i]; }
  return r;
}

// B-matrix 32x16 bf16 fragment per ISA layout:
// lane L: col n = L&15 ; K = (L>>4)*8 + (e&7) + (e>>3)*16
// LDS must hold B as [n][k] (k contiguous).
__device__ __forceinline__ v16bf load_b_frag(const bf16_t* base, int stride,
                                             int n0, int k0, int lane) {
  const bf16_t* p = base + (size_t)(n0 + (lane & 15)) * stride + k0 + ((lane >> 4) << 3);
  v8bf lo = *(const v8bf*)p;
  v8bf hi = *(const v8bf*)(p + 16);
  v16bf r;
#pragma unroll
  for (int i = 0; i < 8; ++i) { r[i] = lo[i]; r[i + 8] = hi[i]; }
  return r;
}

__device__ __forceinline__ v8f wmma_bf16(v16bf a, v16bf b, v8f c) {
  return __builtin_amdgcn_wmma_f32_16x16x32_bf16(false, a, false, b, (short)0, c, false, false);
}

// ---------- kernels ----------

__global__ void convert_f32_bf16(const float* __restrict__ in,
                                 unsigned short* __restrict__ out, int n) {
  int i = blockIdx.x * blockDim.x + threadIdx.x;
  if (i < n) out[i] = f2bf_bits(in[i]);
}

// C = A(bf16, MxK row-major) @ W^T + bias, where W is (N,K) row-major (nn.Linear).
// mode 1: write bf16 to [B,H,T,hd]    (Q,K projections)
// mode 2: write bf16 to [B,H,hd,T]    (V projection, pre-transposed for P@V)
// mode 3: write f32  to [M,N]         (output projection)
__global__ __launch_bounds__(128) void gemm_bf16(
    const unsigned short* __restrict__ A, const unsigned short* __restrict__ W,
    const float* __restrict__ bias, void* __restrict__ out,
    int M, int N, int K, int mode) {
  __shared__ __align__(16) bf16_t As[64 * LDS_STRIDE];
  __shared__ __align__(16) bf16_t Bs[64 * LDS_STRIDE];
  const int tid  = threadIdx.x;
  const int lane = tid & 31;
  const int wave = tid >> 5;
  const int nBase = blockIdx.x * 64;
  const int mBase = blockIdx.y * 64;
  const bf16_t* Ab = (const bf16_t*)A;
  const bf16_t* Wb = (const bf16_t*)W;
  const unsigned asOff = lds_off(As);
  const unsigned bsOff = lds_off(Bs);

  v8f acc[4];
#pragma unroll
  for (int i = 0; i < 4; ++i)
#pragma unroll
    for (int j = 0; j < 8; ++j) acc[i][j] = 0.f;

  for (int k0 = 0; k0 < K; k0 += 64) {
    if (wave == 0) {
      tdm_load_tile64x64(asOff, Ab + (size_t)mBase * K + k0, (unsigned)K, (unsigned)M);
      tdm_load_tile64x64(bsOff, Wb + (size_t)nBase * K + k0, (unsigned)K, (unsigned)N);
      __builtin_amdgcn_s_wait_tensorcnt(0);
    }
    __syncthreads();
    v16bf a0 = load_a_frag(As, LDS_STRIDE, wave * 16, 0, lane);
    v16bf a1 = load_a_frag(As, LDS_STRIDE, wave * 16, 32, lane);
#pragma unroll
    for (int nt = 0; nt < 4; ++nt) {
      v16bf b0 = load_b_frag(Bs, LDS_STRIDE, nt * 16, 0, lane);
      v16bf b1 = load_b_frag(Bs, LDS_STRIDE, nt * 16, 32, lane);
      acc[nt] = wmma_bf16(a0, b0, acc[nt]);
      acc[nt] = wmma_bf16(a1, b1, acc[nt]);
    }
    __syncthreads();
  }

  const int half = lane >> 4;
  const int col  = lane & 15;
#pragma unroll
  for (int nt = 0; nt < 4; ++nt) {
    int n = nBase + nt * 16 + col;
    float bv = bias[n];
#pragma unroll
    for (int r = 0; r < 8; ++r) {
      int m = mBase + wave * 16 + half * 8 + r;
      float v = acc[nt][r] + bv;
      if (mode == 3) {
        ((float*)out)[(size_t)m * N + n] = v;
      } else {
        int b = m / SEQ, t = m - b * SEQ;
        int h = n >> 6, d = n & 63;
        size_t idx = (mode == 1)
          ? ((size_t)(b * NUM_HEADS + h) * SEQ + t) * HEAD_DIM + d
          : ((size_t)(b * NUM_HEADS + h) * HEAD_DIM + d) * SEQ + t;
        ((unsigned short*)out)[idx] = f2bf_bits(v);
      }
    }
  }
}

// Flash attention: one workgroup = 64 query rows of one (b,h); 4 waves, 16 rows each.
__global__ __launch_bounds__(128) void attn_kernel(
    const unsigned short* __restrict__ Q, const unsigned short* __restrict__ Kq,
    const unsigned short* __restrict__ Vt, const unsigned char* __restrict__ pad,
    unsigned short* __restrict__ Aout) {
  __shared__ __align__(16) bf16_t Ks[64 * LDS_STRIDE];         // [t'][d]
  __shared__ __align__(16) bf16_t Vs[64 * LDS_STRIDE];         // [d][t']
  __shared__ __align__(16) bf16_t Ps[4 * 16 * LDS_STRIDE];     // per-wave P [m][t']
  const int tid  = threadIdx.x;
  const int lane = tid & 31;
  const int wave = tid >> 5;
  const int half = lane >> 4;
  const int col  = lane & 15;
  const int qb   = blockIdx.x * 64;
  const int h    = blockIdx.y, b = blockIdx.z;
  const size_t bh = (size_t)(b * NUM_HEADS + h);
  const bf16_t* qp = (const bf16_t*)Q  + bh * SEQ * HEAD_DIM;  // [t][d]
  const bf16_t* kp = (const bf16_t*)Kq + bh * SEQ * HEAD_DIM;  // [t'][d]
  const bf16_t* vp = (const bf16_t*)Vt + bh * HEAD_DIM * SEQ;  // [d][t']
  const unsigned ksOff = lds_off(Ks);
  const unsigned vsOff = lds_off(Vs);

  // Q fragments (row strip of 16, K-dim = head_dim 64 -> two 16x32 A-frags), from global
  v16bf aq0, aq1;
  {
    const bf16_t* p = qp + (size_t)(qb + wave * 16 + col) * HEAD_DIM + (half << 3);
    v8bf l0 = *(const v8bf*)(p);      v8bf h0 = *(const v8bf*)(p + 16);
    v8bf l1 = *(const v8bf*)(p + 32); v8bf h1 = *(const v8bf*)(p + 48);
#pragma unroll
    for (int i = 0; i < 8; ++i) {
      aq0[i] = l0[i]; aq0[i + 8] = h0[i];
      aq1[i] = l1[i]; aq1[i + 8] = h1[i];
    }
  }

  float mrow[8], lrow[8];
  v8f accO[4];
#pragma unroll
  for (int r = 0; r < 8; ++r) { mrow[r] = -1e30f; lrow[r] = 0.f; }
#pragma unroll
  for (int dt = 0; dt < 4; ++dt)
#pragma unroll
    for (int r = 0; r < 8; ++r) accO[dt][r] = 0.f;

  const int nkb = blockIdx.x + 1;   // causal: key blocks 0..diag
  for (int jb = 0; jb < nkb; ++jb) {
    const int kbase = jb * 64;
    if (wave == 0) {
      // K tile: rows t'=kbase..+63, cols d=0..63 of [SEQ][64]
      tdm_load_tile64x64(ksOff, kp + (size_t)kbase * HEAD_DIM, HEAD_DIM, SEQ);
      // V^T tile: rows d=0..63, cols t'=kbase..+63 of [64][SEQ]
      tdm_load_tile64x64(vsOff, vp + kbase, SEQ, HEAD_DIM);
      __builtin_amdgcn_s_wait_tensorcnt(0);
    }
    __syncthreads();

    // S = Q @ K^T  (B-frag over Ks[n=t'][k=d])
    v8f s[4];
#pragma unroll
    for (int ct = 0; ct < 4; ++ct) {
      v8f z;
#pragma unroll
      for (int j = 0; j < 8; ++j) z[j] = 0.f;
      v16bf b0 = load_b_frag(Ks, LDS_STRIDE, ct * 16, 0, lane);
      v16bf b1 = load_b_frag(Ks, LDS_STRIDE, ct * 16, 32, lane);
      z = wmma_bf16(aq0, b0, z);
      z = wmma_bf16(aq1, b1, z);
      s[ct] = z;
    }

    // scale + causal + key-padding mask
#pragma unroll
    for (int ct = 0; ct < 4; ++ct) {
      int kc = kbase + ct * 16 + col;
      bool pm = pad[b * SEQ + kc] != 0;
#pragma unroll
      for (int r = 0; r < 8; ++r) {
        int qr = qb + wave * 16 + half * 8 + r;
        float v = s[ct][r] * 0.125f;
        s[ct][r] = (kc > qr || pm) ? -1e30f : v;
      }
    }

    // online softmax: row stats live in the 16 lanes of one half
    float mnew[8], alpha[8];
#pragma unroll
    for (int r = 0; r < 8; ++r) {
      float mx = fmaxf(fmaxf(s[0][r], s[1][r]), fmaxf(s[2][r], s[3][r]));
#pragma unroll
      for (int off = 8; off >= 1; off >>= 1)
        mx = fmaxf(mx, __shfl_xor(mx, off, 32));
      float mn = fmaxf(mrow[r], mx);
      mnew[r]  = mn;
      alpha[r] = __expf(mrow[r] - mn);
      mrow[r]  = mn;
    }

    // P = exp(S - m), stage per-wave into LDS as [m][t'] for A-frag reuse
    bf16_t* pw = Ps + wave * 16 * LDS_STRIDE;
    unsigned short* pwb = (unsigned short*)pw;
#pragma unroll
    for (int ct = 0; ct < 4; ++ct)
#pragma unroll
      for (int r = 0; r < 8; ++r) {
        float p = __expf(s[ct][r] - mnew[r]);
        s[ct][r] = p;
        pwb[(half * 8 + r) * LDS_STRIDE + ct * 16 + col] = f2bf_bits(p);
      }

#pragma unroll
    for (int r = 0; r < 8; ++r) {
      float sum = s[0][r] + s[1][r] + s[2][r] + s[3][r];
#pragma unroll
      for (int off = 8; off >= 1; off >>= 1)
        sum += __shfl_xor(sum, off, 32);
      lrow[r] = lrow[r] * alpha[r] + sum;
    }
#pragma unroll
    for (int dt = 0; dt < 4; ++dt)
#pragma unroll
      for (int r = 0; r < 8; ++r) accO[dt][r] *= alpha[r];

    // O += P @ V   (A from per-wave LDS P — DS ops are in-order per wave;
    //               B-frag over Vs[n=d][k=t'])
    v16bf ap0 = load_a_frag(pw, LDS_STRIDE, 0, 0, lane);
    v16bf ap1 = load_a_frag(pw, LDS_STRIDE, 0, 32, lane);
#pragma unroll
    for (int dt = 0; dt < 4; ++dt) {
      v16bf b0 = load_b_frag(Vs, LDS_STRIDE, dt * 16, 0, lane);
      v16bf b1 = load_b_frag(Vs, LDS_STRIDE, dt * 16, 32, lane);
      accO[dt] = wmma_bf16(ap0, b0, accO[dt]);
      accO[dt] = wmma_bf16(ap1, b1, accO[dt]);
    }
    __syncthreads();
  }

  // normalize and write bf16 attention output as [B,T,D] for the final GEMM
#pragma unroll
  for (int dt = 0; dt < 4; ++dt)
#pragma unroll
    for (int r = 0; r < 8; ++r) {
      int t = qb + wave * 16 + half * 8 + r;
      float v = accO[dt][r] / lrow[r];
      size_t idx = ((size_t)(b * SEQ + t)) * D_MODEL + h * HEAD_DIM + dt * 16 + col;
      Aout[idx] = f2bf_bits(v);
    }
}

// ---------- launch ----------

extern "C" void kernel_launch(void* const* d_in, const int* in_sizes, int n_in,
                              void* d_out, int out_size, void* d_ws, size_t ws_size,
                              hipStream_t stream) {
  const float* x            = (const float*)d_in[0];
  const unsigned char* pad  = (const unsigned char*)d_in[1];
  const float* Wq = (const float*)d_in[2];
  const float* bq = (const float*)d_in[3];
  const float* Wk = (const float*)d_in[4];
  const float* bk = (const float*)d_in[5];
  const float* Wv = (const float*)d_in[6];
  const float* bv = (const float*)d_in[7];
  const float* Wo = (const float*)d_in[8];
  const float* bo = (const float*)d_in[9];
  float* out = (float*)d_out;

  constexpr size_t SZ_ACT = (size_t)MTOT * D_MODEL * 2;      // 16 MB bf16 activations
  constexpr size_t SZ_W   = (size_t)D_MODEL * D_MODEL * 2;   // 2 MB bf16 weights
  char* ws = (char*)d_ws;
  unsigned short* xb   = (unsigned short*)(ws);
  unsigned short* wqb  = (unsigned short*)(ws + SZ_ACT);
  unsigned short* wkb  = (unsigned short*)(ws + SZ_ACT + SZ_W);
  unsigned short* wvb  = (unsigned short*)(ws + SZ_ACT + 2 * SZ_W);
  unsigned short* wob  = (unsigned short*)(ws + SZ_ACT + 3 * SZ_W);
  unsigned short* qbuf = (unsigned short*)(ws + SZ_ACT + 4 * SZ_W);
  unsigned short* kbuf = (unsigned short*)(ws + 2 * SZ_ACT + 4 * SZ_W);
  unsigned short* vtbf = (unsigned short*)(ws + 3 * SZ_ACT + 4 * SZ_W);
  unsigned short* aout = (unsigned short*)(ws + 4 * SZ_ACT + 4 * SZ_W);

  {
    int n = MTOT * D_MODEL;
    convert_f32_bf16<<<(n + 255) / 256, 256, 0, stream>>>(x, xb, n);
    int nw = D_MODEL * D_MODEL;
    convert_f32_bf16<<<(nw + 255) / 256, 256, 0, stream>>>(Wq, wqb, nw);
    convert_f32_bf16<<<(nw + 255) / 256, 256, 0, stream>>>(Wk, wkb, nw);
    convert_f32_bf16<<<(nw + 255) / 256, 256, 0, stream>>>(Wv, wvb, nw);
    convert_f32_bf16<<<(nw + 255) / 256, 256, 0, stream>>>(Wo, wob, nw);
  }

  dim3 gg(D_MODEL / 64, MTOT / 64);   // (16, 128)
  gemm_bf16<<<gg, 128, 0, stream>>>(xb, wqb, bq, qbuf, MTOT, D_MODEL, D_MODEL, 1);
  gemm_bf16<<<gg, 128, 0, stream>>>(xb, wkb, bk, kbuf, MTOT, D_MODEL, D_MODEL, 1);
  gemm_bf16<<<gg, 128, 0, stream>>>(xb, wvb, bv, vtbf, MTOT, D_MODEL, D_MODEL, 2);

  dim3 ga(SEQ / 64, NUM_HEADS, BATCH);
  attn_kernel<<<ga, 128, 0, stream>>>(qbuf, kbuf, vtbf, pad, aout);

  gemm_bf16<<<gg, 128, 0, stream>>>(aout, wob, bo, out, MTOT, D_MODEL, D_MODEL, 3);
}
